// SSD_72859825209989
// MI455X (gfx1250) — compile-verified, matrix operand order checked by reading, not run on hardware
//
#include <hip/hip_runtime.h>

// SSD post-processing for MI455X (gfx1250).
// Phase 1: decode boxes (trivial).
// Phase 2: softmax over 81 classes; the per-prior exp-sum is computed on the
//          matrix pipe with compensated-f16 v_wmma_f32_16x16x32_f16 (hi+lo
//          split -> ~f32 accuracy); scores written transposed [B,C,P].
// Phase 3: per (b,c): TDM-streamed (tensor_load_to_lds, TENSORcnt) radix
//          top-200 + bitonic sort + bitmask greedy NMS, all in LDS.

#ifndef __has_builtin
#define __has_builtin(x) 0
#endif

#if defined(__AMDGCN__) && __has_builtin(__builtin_amdgcn_tensor_load_to_lds) && \
    __has_builtin(__builtin_amdgcn_s_wait_tensorcnt)
#define USE_TDM 1
#else
#define USE_TDM 0
#endif

namespace {
constexpr int B = 32, P = 24564, C = 81, TOPK = 200;
constexpr float CONF_T = 0.01f;
constexpr float NMS_T = 0.45f;
constexpr int NCAND = 512;                 // candidate pool for final sort
constexpr int TILE = 3072;                 // floats per TDM tile (12 KB)
constexpr int NT = (P + TILE - 1) / TILE;  // 8 tiles
constexpr int PTILES = (P + 15) / 16;      // 1536 16-prior tiles per image
constexpr int BLK_PER_B = PTILES / 8;      // 192 blocks (8 waves each)
}  // namespace

typedef __attribute__((ext_vector_type(16))) _Float16 v16h;
typedef __attribute__((ext_vector_type(8))) float v8f;
typedef __attribute__((ext_vector_type(4))) unsigned int u32x4;
typedef __attribute__((ext_vector_type(8))) int i32x8;
typedef __attribute__((ext_vector_type(4))) int i32x4;

// ---------------------------------------------------------------- decode ----
__global__ __launch_bounds__(256) void ssd_decode(const float* __restrict__ loc,
                                                  const float* __restrict__ priors,
                                                  float* __restrict__ boxes) {
  const int i = blockIdx.x * 256 + threadIdx.x;  // over B*P
  if (i >= B * P) return;
  const int p = i % P;
  const float* l = loc + (size_t)i * 4;
  const float* pr = priors + (size_t)p * 4;
  const float cx = pr[0] + l[0] * 0.1f * pr[2];
  const float cy = pr[1] + l[1] * 0.1f * pr[3];
  const float w = pr[2] * expf(l[2] * 0.2f);
  const float h = pr[3] * expf(l[3] * 0.2f);
  float* o = boxes + (size_t)i * 4;
  o[0] = cx - w * 0.5f;
  o[1] = cy - h * 0.5f;
  o[2] = cx + w * 0.5f;
  o[3] = cy + h * 0.5f;
}

// ------------------------------------------------------- softmax via WMMA ----
// One wave handles 16 priors. A-matrix (16x32 f16) layout per ISA 7.12.2:
// lane L (L<16): row m=L, K = {0..7} in elems 0..7, K = {16..23} in elems 8..15
// lane L (L>=16): row m=L-16, K = {8..15} and {24..31}.
// B = all-ones => D[m][n] = sum_k A[m][k]. Three 32-class chunks cover C=81
// (pad with 0). Compensated hi/lo f16 split -> f32-accurate sums.
__global__ __launch_bounds__(256) void ssd_softmax_wmma(const float* __restrict__ conf,
                                                        float* __restrict__ scoresT) {
  const int wave = threadIdx.x >> 5;
  const int lane = threadIdx.x & 31;
  const int m = lane & 15;
  const int half = lane >> 4;
  const int b = blockIdx.x / BLK_PER_B;
  const int tile = (blockIdx.x % BLK_PER_B) * 8 + wave;
  const int p = tile * 16 + m;
  const int pc = (p < P) ? p : (P - 1);  // clamp loads for tail rows
  const float* __restrict__ row = conf + ((size_t)b * P + pc) * C;

  float e[48];
  float mx = -INFINITY;
#pragma unroll
  for (int ch = 0; ch < 3; ++ch)
#pragma unroll
    for (int sg = 0; sg < 2; ++sg)
#pragma unroll
      for (int j = 0; j < 8; ++j) {
        const int k = ch * 32 + sg * 16 + half * 8 + j;
        const float x = (k < C) ? row[k] : -INFINITY;
        e[ch * 16 + sg * 8 + j] = x;
        mx = fmaxf(mx, x);
      }
  // other half of this row lives in lane^16
  mx = fmaxf(mx, __shfl_xor(mx, 16, 32));

#pragma unroll
  for (int ch = 0; ch < 3; ++ch)
#pragma unroll
    for (int sg = 0; sg < 2; ++sg)
#pragma unroll
      for (int j = 0; j < 8; ++j) {
        const int k = ch * 32 + sg * 16 + half * 8 + j;
        const int idx = ch * 16 + sg * 8 + j;
        e[idx] = (k < C) ? __expf(e[idx] - mx) : 0.0f;
      }

  v16h ones;
#pragma unroll
  for (int i = 0; i < 16; ++i) ones[i] = (_Float16)1.0f;

  v8f acc = {};
#pragma unroll
  for (int ch = 0; ch < 3; ++ch) {
    v16h hi, lo;
#pragma unroll
    for (int i = 0; i < 16; ++i) {
      const float x = e[ch * 16 + i];
      const _Float16 h = (_Float16)x;
      hi[i] = h;
      lo[i] = (_Float16)(x - (float)h);  // residual term
    }
    acc = __builtin_amdgcn_wmma_f32_16x16x32_f16(false, hi, false, ones, (short)0, acc,
                                                 false, false);
    acc = __builtin_amdgcn_wmma_f32_16x16x32_f16(false, lo, false, ones, (short)0, acc,
                                                 false, false);
  }

  // D layout: lane0 holds rows 0..7 (col 0), lane16 holds rows 8..15.
  __shared__ float s_sums[8][16];
  if (lane == 0) {
#pragma unroll
    for (int r = 0; r < 8; ++r) s_sums[wave][r] = acc[r];
  } else if (lane == 16) {
#pragma unroll
    for (int r = 0; r < 8; ++r) s_sums[wave][8 + r] = acc[r];
  }
  __syncthreads();
  const float inv = 1.0f / s_sums[wave][m];

  if (p < P) {
#pragma unroll
    for (int ch = 0; ch < 3; ++ch)
#pragma unroll
      for (int sg = 0; sg < 2; ++sg)
#pragma unroll
        for (int j = 0; j < 8; ++j) {
          const int k = ch * 32 + sg * 16 + half * 8 + j;
          if (k < C)
            scoresT[((size_t)b * C + k) * P + p] = e[ch * 16 + sg * 8 + j] * inv;
        }
  }
}

// ------------------------------------------------ TDM streaming machinery ----
#if USE_TDM
__device__ inline void tdm_issue(const float* gsrc, float* lds_dst, int count) {
  // 2D descriptor: 1 row of `count` 4-byte elements (ISA 08_async_tensor §8).
  const unsigned long long ga = (unsigned long long)(uintptr_t)gsrc;
  const unsigned lds = (unsigned)(uintptr_t)lds_dst;  // flat->LDS = low 32 bits
  const unsigned cnt = (unsigned)count;
  u32x4 g0;
  g0[0] = 1u;                                                 // count=1, user mode
  g0[1] = lds;                                                // lds_addr
  g0[2] = (unsigned)(ga & 0xFFFFFFFFull);                     // global_addr[31:0]
  g0[3] = (unsigned)((ga >> 32) & 0x1FFFFFFull) | (2u << 30); // addr[56:32] | type=2
  i32x8 g1;
  g1[0] = (int)(2u << 16);                 // data_size = 4B
  g1[1] = (int)((cnt & 0xFFFFu) << 16);    // tensor_dim0[15:0]
  g1[2] = (int)((cnt >> 16) | (1u << 16)); // tensor_dim0[31:16], tensor_dim1=1
  g1[3] = (int)((cnt & 0xFFFFu) << 16);    // tile_dim0 = count
  g1[4] = 1;                               // tile_dim1 = 1
  g1[5] = (int)cnt;                        // tensor_dim0_stride
  g1[6] = 0;
  g1[7] = 0;
  i32x4 z4;
  z4[0] = 0; z4[1] = 0; z4[2] = 0; z4[3] = 0;
  i32x8 z8;
  z8[0] = 0; z8[1] = 0; z8[2] = 0; z8[3] = 0;
  z8[4] = 0; z8[5] = 0; z8[6] = 0; z8[7] = 0;
  // 6-arg toolchain variant: (g0, g1, g2, g3, extra, cpol)
  __builtin_amdgcn_tensor_load_to_lds(g0, g1, z4, z4, z8, 0);
}
#endif

// Stream scoresT[b][c][0..P) through double-buffered LDS tiles; f(i, score).
template <typename F>
__device__ void stream_scores(const float* __restrict__ src, float* b0, float* b1, F f) {
#if USE_TDM
  const bool w0 = (threadIdx.x >> 5) == 0;  // one wave drives the TDM
  if (w0) tdm_issue(src, b0, (P < TILE) ? P : TILE);
  for (int t = 0; t < NT; ++t) {
    float* cur = (t & 1) ? b1 : b0;
    float* nxt = (t & 1) ? b0 : b1;
    if (w0) {
      if (t + 1 < NT) {
        const int nb = (t + 1) * TILE;
        const int nc = ((P - nb) < TILE) ? (P - nb) : TILE;
        tdm_issue(src + nb, nxt, nc);
        __builtin_amdgcn_s_wait_tensorcnt(1);  // tile t landed
      } else {
        __builtin_amdgcn_s_wait_tensorcnt(0);
      }
    }
    __syncthreads();
    const int base = t * TILE;
    const int cnt = ((P - base) < TILE) ? (P - base) : TILE;
    for (int i = threadIdx.x; i < cnt; i += 256) f(base + i, cur[i]);
    __syncthreads();
  }
#else
  __syncthreads();
  for (int i = threadIdx.x; i < P; i += 256) f(i, src[i]);
  __syncthreads();
  (void)b0;
  (void)b1;
#endif
}

// Suffix-scan a 4096-bucket histogram, find the bucket of the K-th largest.
// s_res = {cut_bucket, count_strictly_above, total}; total<=K => take-all.
__device__ void find_cut(const unsigned* s_hist, unsigned* s_seg, unsigned* s_res,
                         unsigned K) {
  const int tid = threadIdx.x;
  unsigned s = 0;
#pragma unroll 4
  for (int i = 0; i < 16; ++i) s += s_hist[tid * 16 + i];
  s_seg[tid] = s;
  __syncthreads();
  for (int off = 1; off < 256; off <<= 1) {  // inclusive suffix scan
    const unsigned v = (tid + off < 256) ? s_seg[tid + off] : 0u;
    __syncthreads();
    s_seg[tid] += v;
    __syncthreads();
  }
  if (tid == 0) {
    const unsigned total = s_seg[0];
    if (total <= K) {
      s_res[0] = 0u;
      s_res[1] = 0u;
      s_res[2] = total;
    } else {
      int t = 255;
      while (t > 0 && s_seg[t] < K) --t;
      unsigned acc = (t < 255) ? s_seg[t + 1] : 0u;
      unsigned cut = (unsigned)(t * 16), above = acc;
      for (int i = 15; i >= 0; --i) {
        const unsigned h = s_hist[t * 16 + i];
        if (acc + h >= K) {
          cut = (unsigned)(t * 16 + i);
          above = acc;
          break;
        }
        acc += h;
      }
      s_res[0] = cut;
      s_res[1] = above;
      s_res[2] = total;
    }
  }
  __syncthreads();
}

// ---------------------------------------------------------- top-k + NMS ----
__global__ __launch_bounds__(256) void ssd_topk_nms(const float* __restrict__ scoresT,
                                                    const float* __restrict__ boxes,
                                                    float* __restrict__ out) {
  __shared__ float s_buf0[TILE];
  __shared__ float s_buf1[TILE];
  __shared__ unsigned s_hist[4096];
  __shared__ unsigned s_seg[256];
  __shared__ unsigned s_res[3];
  __shared__ unsigned s_res2[3];
  __shared__ unsigned s_count;
  __shared__ unsigned long long s_keys[NCAND];
  __shared__ float s_bx[TOPK * 4];
  __shared__ float s_sc[TOPK];
  __shared__ float s_area[TOPK];
  __shared__ unsigned s_sup[TOPK * 7];
  __shared__ unsigned s_keep[7];

  const int tid = threadIdx.x;
  const int b = blockIdx.x / C;
  const int c = blockIdx.x % C;
  const float* __restrict__ src = scoresT + ((size_t)b * C + c) * P;

  // Pass 1: histogram on float bits[30:19] (exp + 4 mantissa bits).
  for (int i = tid; i < 4096; i += 256) s_hist[i] = 0u;
  stream_scores(src, s_buf0, s_buf1, [&](int i, float s) {
    (void)i;
    if (s > CONF_T) atomicAdd(&s_hist[(__float_as_uint(s) >> 19) & 0xFFFu], 1u);
  });
  find_cut(s_hist, s_seg, s_res, (unsigned)TOPK);
  const unsigned cutB = s_res[0];
  const unsigned above = s_res[1];
  const bool allpass = (s_res[2] <= (unsigned)TOPK);  // block-uniform

  // Pass 2: refine cut bucket on bits[18:7].
  unsigned subCut = 0u;
  if (!allpass) {
    for (int i = tid; i < 4096; i += 256) s_hist[i] = 0u;
    stream_scores(src, s_buf0, s_buf1, [&](int i, float s) {
      (void)i;
      if (s > CONF_T) {
        const unsigned u = __float_as_uint(s);
        if (((u >> 19) & 0xFFFu) == cutB) atomicAdd(&s_hist[(u >> 7) & 0xFFFu], 1u);
      }
    });
    find_cut(s_hist, s_seg, s_res2, (unsigned)TOPK - above);
    subCut = s_res2[0];
  }

  // Pass 3: collect candidates as 64-bit keys (score bits || ~index) so a
  // descending sort reproduces top_k's value-desc / index-asc order.
  if (tid == 0) s_count = 0u;
  stream_scores(src, s_buf0, s_buf1, [&](int i, float s) {
    if (s > CONF_T) {
      const unsigned u = __float_as_uint(s);
      const unsigned bkt = (u >> 19) & 0xFFFu;
      bool take = allpass || (bkt > cutB);
      if (!take && bkt == cutB) take = ((u >> 7) & 0xFFFu) >= subCut;
      if (take) {
        const unsigned idx = atomicAdd(&s_count, 1u);
        if (idx < (unsigned)NCAND)
          s_keys[idx] = ((unsigned long long)u << 32) | (unsigned)(~(unsigned)i);
      }
    }
  });
  const unsigned n = (s_count < (unsigned)NCAND) ? s_count : (unsigned)NCAND;
  for (int i = tid; i < NCAND; i += 256)
    if ((unsigned)i >= n) s_keys[i] = 0ull;
  __syncthreads();

  // Bitonic sort, descending.
  for (int k = 2; k <= NCAND; k <<= 1)
    for (int j = k >> 1; j > 0; j >>= 1) {
      for (int i = tid; i < NCAND; i += 256) {
        const int ixj = i ^ j;
        if (ixj > i) {
          const bool desc = ((i & k) == 0);
          const unsigned long long a = s_keys[i], bb = s_keys[ixj];
          if (desc ? (a < bb) : (a > bb)) {
            s_keys[i] = bb;
            s_keys[ixj] = a;
          }
        }
      }
      __syncthreads();
    }

  // Gather top-200 boxes.
  if (tid < TOPK) {
    const unsigned long long key = s_keys[tid];
    const unsigned ub = (unsigned)(key >> 32);
    const unsigned p = ~((unsigned)(key & 0xFFFFFFFFull));
    float x1 = 0.f, y1 = 0.f, x2 = 0.f, y2 = 0.f, sc = 0.f;
    if (key != 0ull && p < (unsigned)P) {
      sc = __uint_as_float(ub);
      const float* bx = boxes + ((size_t)b * P + p) * 4;
      x1 = bx[0];
      y1 = bx[1];
      x2 = bx[2];
      y2 = bx[3];
    }
    s_sc[tid] = sc;
    s_bx[tid * 4 + 0] = x1;
    s_bx[tid * 4 + 1] = y1;
    s_bx[tid * 4 + 2] = x2;
    s_bx[tid * 4 + 3] = y2;
    s_area[tid] = fmaxf(x2 - x1, 0.f) * fmaxf(y2 - y1, 0.f);
  }
  __syncthreads();

  // Suppression bitmasks: sup[i] bit j set iff j>i and IoU(i,j) > NMS_T.
  for (int w = tid; w < TOPK * 7; w += 256) {
    const int i = w / 7, wj = w % 7;
    const float ax1 = s_bx[i * 4], ay1 = s_bx[i * 4 + 1];
    const float ax2 = s_bx[i * 4 + 2], ay2 = s_bx[i * 4 + 3];
    const float ai = s_area[i];
    unsigned mask = 0u;
    for (int bit = 0; bit < 32; ++bit) {
      const int j = wj * 32 + bit;
      if (j < TOPK && j > i) {
        const float lx = fmaxf(ax1, s_bx[j * 4]);
        const float ly = fmaxf(ay1, s_bx[j * 4 + 1]);
        const float rx = fminf(ax2, s_bx[j * 4 + 2]);
        const float ry = fminf(ay2, s_bx[j * 4 + 3]);
        const float iw = fmaxf(rx - lx, 0.f), ih = fmaxf(ry - ly, 0.f);
        const float inter = iw * ih;
        const float iou = inter / (ai + s_area[j] - inter + 1e-9f);
        if (iou > NMS_T) mask |= (1u << bit);
      }
    }
    s_sup[i * 7 + wj] = mask;
  }
  __syncthreads();

  // Greedy suppression over bitmasks (1400 word-ANDs, single lane).
  if (tid == 0) {
    unsigned keep[7] = {0xFFFFFFFFu, 0xFFFFFFFFu, 0xFFFFFFFFu, 0xFFFFFFFFu,
                        0xFFFFFFFFu, 0xFFFFFFFFu, 0xFFu};
    for (int i = 0; i < TOPK; ++i)
      if ((keep[i >> 5] >> (i & 31)) & 1u)
        for (int w = 0; w < 7; ++w) keep[w] &= ~s_sup[i * 7 + w];
    for (int w = 0; w < 7; ++w) s_keep[w] = keep[w];
  }
  __syncthreads();

  if (tid < TOPK) {
    const float kf = ((s_keep[tid >> 5] >> (tid & 31)) & 1u) ? 1.0f : 0.0f;
    float* o = out + (((size_t)b * C + c) * TOPK + tid) * 5;
    o[0] = s_sc[tid] * kf;
    o[1] = s_bx[tid * 4 + 0];
    o[2] = s_bx[tid * 4 + 1];
    o[3] = s_bx[tid * 4 + 2];
    o[4] = s_bx[tid * 4 + 3];
  }
}

// ------------------------------------------------------------------ host ----
extern "C" void kernel_launch(void* const* d_in, const int* in_sizes, int n_in,
                              void* d_out, int out_size, void* d_ws, size_t ws_size,
                              hipStream_t stream) {
  const float* loc = (const float*)d_in[0];     // [B,P,4]
  const float* conf = (const float*)d_in[1];    // [B,P,C]
  const float* priors = (const float*)d_in[2];  // [P,4]
  float* out = (float*)d_out;                   // [B,C,TOPK,5]

  // Workspace: transposed scores [B,C,P] then decoded boxes [B,P,4].
  float* scoresT = (float*)d_ws;
  float* boxes = scoresT + (size_t)B * C * P;

  ssd_decode<<<(B * P + 255) / 256, 256, 0, stream>>>(loc, priors, boxes);
  ssd_softmax_wmma<<<B * BLK_PER_B, 256, 0, stream>>>(conf, scoresT);
  ssd_topk_nms<<<B * C, 256, 0, stream>>>(scoresT, boxes, out);

  (void)in_sizes;
  (void)n_in;
  (void)out_size;
  (void)ws_size;
}
